// MultiStepDynamicsModel_8229157339352
// MI455X (gfx1250) — matrix operand order repaired and checked by reading, not compiled, not run on hardware
//
#include <hip/hip_runtime.h>
#include <stdint.h>

// ---------------------------------------------------------------------------
// MultiStepDynamicsModel for MI455X (gfx1250): bf16 WMMA GEMMs + fused eltwise
// B=2048, T=32, OBS=128, ACT=32, HID=1024, LAT=512, L=2
// Weights pre-converted to bf16 AND pre-transposed (W^T [N][K]) once per
// launch so both GEMM tiles stage into LDS as contiguous 16B chunks via the
// gfx1250 async global->LDS path (ASYNCcnt), double-buffered in LDS.
// ---------------------------------------------------------------------------

typedef __bf16 bf16;
typedef __attribute__((ext_vector_type(16))) __bf16 v16bf;
typedef __attribute__((ext_vector_type(8)))  __bf16 v8bf;
typedef __attribute__((ext_vector_type(8)))  float   v8f;

#define BD   2048
#define TT   32
#define OBSD 128
#define ACTD 32
#define HIDD 1024
#define LATD 512

#if defined(__has_builtin)
#  if __has_builtin(__builtin_amdgcn_global_load_async_to_lds_b128)
#    define USE_ASYNC_LDS 1
#  endif
#endif

#ifdef USE_ASYNC_LDS
typedef int v4i __attribute__((vector_size(16)));
typedef __attribute__((address_space(1))) v4i* gv4_t;   // global int4*
typedef __attribute__((address_space(3))) v4i* lv4_t;   // LDS int4*
__device__ __forceinline__ void async_cp16(const void* g, void* l)
{
    // global_load_async_to_lds_b128: 16B per lane, tracked by ASYNCcnt
    __builtin_amdgcn_global_load_async_to_lds_b128(
        (gv4_t)(uintptr_t)g,
        (lv4_t)(unsigned int)(uintptr_t)l, 0, 0);
}
#endif

// ---------------------------------------------------------------------------
// GEMM:  out = act( A[M,K] @ W[K,N] + bias[N] (+ Cadd[M,N]) ), W given as
// WT[N][K] row-major. Block tile 128x128, K-step 32, 256 threads = 8 wave32
// (4 in M x 2 in N), each wave owns 2x4 tiles of v_wmma_f32_16x16x32_bf16.
// ---------------------------------------------------------------------------
__global__ __launch_bounds__(256)
void gemm_bf16(const bf16* __restrict__ A, int lda,
               const bf16* __restrict__ WT, int ldw,
               const float* __restrict__ bias,
               const float* __restrict__ Cadd, int ldadd,
               float* __restrict__ outF, int ldf,
               bf16* __restrict__ outB, int ldb,
               int K, int act)
{
    // double-buffered tiles; rows padded to 40 bf16 (80B = 20 dwords:
    // 16 lanes reading rows r..r+15 hit 16 distinct banks)
    __shared__ __align__(16) bf16 lA[2][128 * 40];
    __shared__ __align__(16) bf16 lB[2][128 * 40];

    const int tid   = threadIdx.x;
    const int lane  = tid & 31;
    const int w     = tid >> 5;
    const int lh    = lane & 15;
    const int hi    = lane >> 4;
    const int wmRow = (w & 3) * 32;   // 4 waves tile M: 32 rows each
    const int wnCol = (w >> 2) * 64;  // 2 waves tile N: 64 cols each
    const int m0    = blockIdx.y * 128;
    const int n0    = blockIdx.x * 128;

    // staging: thread -> tile row (tid>>1), k-half (tid&1)*16 (2 x 16B)
    const int srow = tid >> 1;
    const int skq  = (tid & 1) * 16;
    const bf16* Ag = A  + (size_t)(m0 + srow) * lda + skq;
    const bf16* Wg = WT + (size_t)(n0 + srow) * ldw + skq;
    bf16* lAs = &lA[0][srow * 40 + skq];
    bf16* lBs = &lB[0][srow * 40 + skq];
    const int BUFO = 128 * 40;

    const int nst = K >> 5;

    // ---- prologue: stage s=0 into buffer 0 ----
#ifdef USE_ASYNC_LDS
    async_cp16(Ag,     lAs);
    async_cp16(Ag + 8, lAs + 8);
    async_cp16(Wg,     lBs);
    async_cp16(Wg + 8, lBs + 8);
    asm volatile("s_wait_asynccnt 0x0" ::: "memory");
#else
    {
        uint4 a0 = *(const uint4*)Ag, a1 = *(const uint4*)(Ag + 8);
        uint4 b0 = *(const uint4*)Wg, b1 = *(const uint4*)(Wg + 8);
        *(uint4*)lAs = a0; *(uint4*)(lAs + 8) = a1;
        *(uint4*)lBs = b0; *(uint4*)(lBs + 8) = b1;
    }
#endif
    __syncthreads();

    v8f acc[2][4] = {};

    for (int s = 0; s < nst; ++s) {
        const int  cur = s & 1;
        const int  nxt = cur ^ 1;
        const bool has = (s + 1) < nst;
        const int  kn  = (s + 1) << 5;

#ifndef USE_ASYNC_LDS
        uint4 a0, a1, b0, b1;
#endif
        if (has) {  // issue next-stage staging before compute (overlap)
#ifdef USE_ASYNC_LDS
            async_cp16(Ag + kn,     lAs + nxt * BUFO);
            async_cp16(Ag + kn + 8, lAs + nxt * BUFO + 8);
            async_cp16(Wg + kn,     lBs + nxt * BUFO);
            async_cp16(Wg + kn + 8, lBs + nxt * BUFO + 8);
#else
            a0 = *(const uint4*)(Ag + kn); a1 = *(const uint4*)(Ag + kn + 8);
            b0 = *(const uint4*)(Wg + kn); b1 = *(const uint4*)(Wg + kn + 8);
#endif
        }
        if (s + 2 < nst) {  // L2 hint for stage s+2 -> global_prefetch_b8
            __builtin_prefetch(Ag + ((s + 2) << 5), 0, 1);
            __builtin_prefetch(Wg + ((s + 2) << 5), 0, 1);
        }

        // ---- fragments from buffer `cur` ----
        const bf16* pA = lA[cur];
        const bf16* pB = lB[cur];
        // A 16-bit layout: lane lo-half K=[0..7],[16..23]; hi-half K=[8..15],[24..31]
        v16bf af[2];
        #pragma unroll
        for (int tm = 0; tm < 2; ++tm) {
            int r = wmRow + tm * 16 + lh;
            ((v8bf*)&af[tm])[0] = *(const v8bf*)&pA[r * 40 + hi * 8];
            ((v8bf*)&af[tm])[1] = *(const v8bf*)&pA[r * 40 + 16 + hi * 8];
        }
        // B 16-bit layout: lane holds column n, lo-half K=[0..7],[16..23]; hi-half K=[8..15],[24..31]
        v16bf bfr[4];
        #pragma unroll
        for (int tn = 0; tn < 4; ++tn) {
            int cn = wnCol + tn * 16 + lh;
            ((v8bf*)&bfr[tn])[0] = *(const v8bf*)&pB[cn * 40 + hi * 8];
            ((v8bf*)&bfr[tn])[1] = *(const v8bf*)&pB[cn * 40 + 16 + hi * 8];
        }

        #pragma unroll
        for (int tm = 0; tm < 2; ++tm)
            #pragma unroll
            for (int tn = 0; tn < 4; ++tn)
                acc[tm][tn] = __builtin_amdgcn_wmma_f32_16x16x32_bf16(
                    false, af[tm], false, bfr[tn], (short)0, acc[tm][tn], false, false);

        if (has) {
#ifdef USE_ASYNC_LDS
            asm volatile("s_wait_asynccnt 0x0" ::: "memory");
#else
            *(uint4*)(lAs + nxt * BUFO)     = a0;
            *(uint4*)(lAs + nxt * BUFO + 8) = a1;
            *(uint4*)(lBs + nxt * BUFO)     = b0;
            *(uint4*)(lBs + nxt * BUFO + 8) = b1;
#endif
        }
        __syncthreads();
    }

    // ---- epilogue: C layout row = r + hi*8, col = lh ----
    #pragma unroll
    for (int tm = 0; tm < 2; ++tm) {
        #pragma unroll
        for (int tn = 0; tn < 4; ++tn) {
            int colg = n0 + wnCol + tn * 16 + lh;
            float bv = bias ? bias[colg] : 0.0f;
            #pragma unroll
            for (int r = 0; r < 8; ++r) {
                int rowg = m0 + wmRow + tm * 16 + hi * 8 + r;
                float v = acc[tm][tn][r] + bv;
                if (Cadd) v += Cadd[(size_t)rowg * ldadd + colg];
                if (act)  v = fmaxf(v, 0.0f);
                if (outF) outF[(size_t)rowg * ldf + colg] = v;
                if (outB) outB[(size_t)rowg * ldb + colg] = (bf16)v;
            }
        }
    }
}

// ---------------------------------------------------------------------------
// Elementwise helpers
// ---------------------------------------------------------------------------
__global__ void concat_xu(const float* __restrict__ x, const float* __restrict__ u,
                          bf16* __restrict__ xu, int n)
{
    int i = blockIdx.x * blockDim.x + threadIdx.x;
    if (i >= n) return;
    int row = i / (OBSD + ACTD), col = i % (OBSD + ACTD);
    float v = (col < OBSD) ? x[(size_t)row * OBSD + col]
                           : u[(size_t)row * ACTD + (col - OBSD)];
    xu[i] = (bf16)v;
}

__device__ __forceinline__ float sigf(float x) { return 1.0f / (1.0f + expf(-x)); }

__global__ void lstm_gate(const float* __restrict__ z, float* __restrict__ c,
                          bf16* __restrict__ h0, int ld0,
                          bf16* __restrict__ h1, int ld1,
                          float* __restrict__ hF, int n)
{
    int idx = blockIdx.x * blockDim.x + threadIdx.x;
    if (idx >= n) return;
    int row = idx / LATD, col = idx % LATD;
    const float* zr = z + (size_t)row * (4 * LATD);
    float ig = sigf(zr[col]);
    float fg = sigf(zr[LATD + col]);
    float gg = tanhf(zr[2 * LATD + col]);
    float og = sigf(zr[3 * LATD + col]);
    float cn = fg * c[idx] + ig * gg;
    c[idx] = cn;
    float h = og * tanhf(cn);
    h0[(size_t)row * ld0 + col] = (bf16)h;
    if (h1) h1[(size_t)row * ld1 + col] = (bf16)h;
    if (hF) hF[idx] = h;
}

// d[n*ldd + k] = (bf16)s[k*N + n]  (convert f32->bf16 + transpose, prep only)
__global__ void cvtT(const float* __restrict__ s, bf16* __restrict__ d,
                     int K, int N, int ldd)
{
    int i = blockIdx.x * blockDim.x + threadIdx.x;
    if (i >= K * N) return;
    int k = i / N, n = i % N;
    d[(size_t)n * ldd + k] = (bf16)s[i];
}

__global__ void bias_sum(const float* __restrict__ a, const float* __restrict__ b,
                         float* __restrict__ d, int n)
{
    int i = blockIdx.x * blockDim.x + threadIdx.x;
    if (i < n) d[i] = a[i] + b[i];
}

__global__ void zero_u32(unsigned int* __restrict__ p, int nwords)
{
    int i = blockIdx.x * blockDim.x + threadIdx.x;
    if (i < nwords) p[i] = 0u;
}

// ---------------------------------------------------------------------------
// Host-side orchestration
// ---------------------------------------------------------------------------
static inline int gdiv(int a, int b) { return (a + b - 1) / b; }

extern "C" void kernel_launch(void* const* d_in, const int* in_sizes, int n_in,
                              void* d_out, int out_size, void* d_ws, size_t ws_size,
                              hipStream_t stream)
{
    (void)in_sizes; (void)n_in; (void)out_size; (void)ws_size;

    const float* x_in = (const float*)d_in[0];
    const float* us   = (const float*)d_in[1];
    const float* eW0  = (const float*)d_in[2];  const float* eb0 = (const float*)d_in[3];
    const float* eW1  = (const float*)d_in[4];  const float* eb1 = (const float*)d_in[5];
    const float* eW2  = (const float*)d_in[6];  const float* eb2 = (const float*)d_in[7];
    const float* dW0  = (const float*)d_in[8];  const float* db0 = (const float*)d_in[9];
    const float* dW1  = (const float*)d_in[10]; const float* db1 = (const float*)d_in[11];
    const float* dW2  = (const float*)d_in[12]; const float* db2 = (const float*)d_in[13];
    const float* lWi  = (const float*)d_in[14];
    const float* lWh  = (const float*)d_in[15];
    const float* lbi  = (const float*)d_in[16];
    const float* lbh  = (const float*)d_in[17];

    float* outF = (float*)d_out;                       // [T,B,OBS] then [B,LAT]
    float* hfin = outF + (size_t)TT * BD * OBSD;

    char*  ws = (char*)d_ws;
    size_t o  = 0;
    auto sub = [&](size_t bytes) -> char* {
        char* p = ws + o;
        o += (bytes + 255) & ~(size_t)255;
        return p;
    };
    // transposed bf16 weights: WT[N][K]
    bf16* encW0T = (bf16*)sub((size_t)HIDD * (OBSD + ACTD) * 2);
    bf16* encW1T = (bf16*)sub((size_t)HIDD * HIDD * 2);
    bf16* encW2T = (bf16*)sub((size_t)LATD * HIDD * 2);
    bf16* wcat0T = (bf16*)sub((size_t)(4 * LATD) * (2 * LATD) * 2); // [Wi0;Wh0]^T
    bf16* wcat1T = (bf16*)sub((size_t)(4 * LATD) * (2 * LATD) * 2); // [Wi1;Wh1]^T
    bf16* decW0T = (bf16*)sub((size_t)HIDD * LATD * 2);
    bf16* decW1T = (bf16*)sub((size_t)HIDD * HIDD * 2);
    bf16* decW2T = (bf16*)sub((size_t)OBSD * HIDD * 2);
    float* bcat  = (float*)sub((size_t)2 * 4 * LATD * 4);           // bi+bh per layer
    bf16* XU     = (bf16*)sub((size_t)BD * (OBSD + ACTD) * 2);
    bf16* E0     = (bf16*)sub((size_t)BD * HIDD * 2);
    bf16* E1     = (bf16*)sub((size_t)BD * HIDD * 2);
    bf16* L0in   = (bf16*)sub((size_t)BD * (2 * LATD) * 2);         // [inp | h0]
    bf16* L1in   = (bf16*)sub((size_t)BD * (2 * LATD) * 2);         // [h0  | h1]
    float* Zg    = (float*)sub((size_t)BD * (4 * LATD) * 4);
    float* c0    = (float*)sub((size_t)BD * LATD * 4);
    float* c1    = (float*)sub((size_t)BD * LATD * 4);

    const int TPB = 256;
    const int Wsl = LATD * (4 * LATD); // one lstm weight matrix (512x2048)

    // ---- per-launch prep: convert+transpose weights, fuse biases, zero state
    cvtT<<<gdiv((OBSD + ACTD) * HIDD, TPB), TPB, 0, stream>>>(eW0, encW0T, OBSD + ACTD, HIDD, OBSD + ACTD);
    cvtT<<<gdiv(HIDD * HIDD, TPB), TPB, 0, stream>>>(eW1, encW1T, HIDD, HIDD, HIDD);
    cvtT<<<gdiv(HIDD * LATD, TPB), TPB, 0, stream>>>(eW2, encW2T, HIDD, LATD, HIDD);
    cvtT<<<gdiv(Wsl, TPB), TPB, 0, stream>>>(lWi,       wcat0T,        LATD, 4 * LATD, 2 * LATD);
    cvtT<<<gdiv(Wsl, TPB), TPB, 0, stream>>>(lWh,       wcat0T + LATD, LATD, 4 * LATD, 2 * LATD);
    cvtT<<<gdiv(Wsl, TPB), TPB, 0, stream>>>(lWi + Wsl, wcat1T,        LATD, 4 * LATD, 2 * LATD);
    cvtT<<<gdiv(Wsl, TPB), TPB, 0, stream>>>(lWh + Wsl, wcat1T + LATD, LATD, 4 * LATD, 2 * LATD);
    cvtT<<<gdiv(LATD * HIDD, TPB), TPB, 0, stream>>>(dW0, decW0T, LATD, HIDD, LATD);
    cvtT<<<gdiv(HIDD * HIDD, TPB), TPB, 0, stream>>>(dW1, decW1T, HIDD, HIDD, HIDD);
    cvtT<<<gdiv(HIDD * OBSD, TPB), TPB, 0, stream>>>(dW2, decW2T, HIDD, OBSD, HIDD);
    bias_sum<<<gdiv(2 * 4 * LATD, TPB), TPB, 0, stream>>>(lbi, lbh, bcat, 2 * 4 * LATD);
    zero_u32<<<gdiv(BD * LATD, TPB), TPB, 0, stream>>>((unsigned int*)c0, BD * LATD);
    zero_u32<<<gdiv(BD * LATD, TPB), TPB, 0, stream>>>((unsigned int*)c1, BD * LATD);
    zero_u32<<<gdiv(BD * LATD, TPB), TPB, 0, stream>>>((unsigned int*)L0in, BD * LATD); // bf16: 2*LAT cols /2
    zero_u32<<<gdiv(BD * LATD, TPB), TPB, 0, stream>>>((unsigned int*)L1in, BD * LATD);

    const dim3 blk(TPB);
    const dim3 gHID(HIDD / 128, BD / 128);
    const dim3 gLAT(LATD / 128, BD / 128);
    const dim3 gGATE(4 * LATD / 128, BD / 128);
    const dim3 gOBS(OBSD / 128, BD / 128);

    for (int t = 0; t < TT; ++t) {
        const float* xt = (t == 0) ? x_in : (outF + (size_t)(t - 1) * BD * OBSD);
        const float* ut = us + (size_t)t * BD * ACTD;

        concat_xu<<<gdiv(BD * (OBSD + ACTD), TPB), blk, 0, stream>>>(xt, ut, XU, BD * (OBSD + ACTD));

        // ---- encoder ----
        gemm_bf16<<<gHID, blk, 0, stream>>>(XU, OBSD + ACTD, encW0T, OBSD + ACTD, eb0,
                                            nullptr, 0, nullptr, 0, E0, HIDD, OBSD + ACTD, 1);
        gemm_bf16<<<gHID, blk, 0, stream>>>(E0, HIDD, encW1T, HIDD, eb1,
                                            nullptr, 0, nullptr, 0, E1, HIDD, HIDD, 1);
        gemm_bf16<<<gLAT, blk, 0, stream>>>(E1, HIDD, encW2T, HIDD, eb2,
                                            nullptr, 0, nullptr, 0, L0in, 2 * LATD, HIDD, 0);

        // ---- LSTM layer 0: z = [inp|h0] @ [Wi0;Wh0] + (bi0+bh0) ----
        gemm_bf16<<<gGATE, blk, 0, stream>>>(L0in, 2 * LATD, wcat0T, 2 * LATD, bcat,
                                             nullptr, 0, Zg, 4 * LATD, nullptr, 0, 2 * LATD, 0);
        lstm_gate<<<gdiv(BD * LATD, TPB), blk, 0, stream>>>(Zg, c0,
                                             L0in + LATD, 2 * LATD,   // h0 for next step
                                             L1in, 2 * LATD,          // input to layer 1
                                             nullptr, BD * LATD);

        // ---- LSTM layer 1 ----
        gemm_bf16<<<gGATE, blk, 0, stream>>>(L1in, 2 * LATD, wcat1T, 2 * LATD, bcat + 4 * LATD,
                                             nullptr, 0, Zg, 4 * LATD, nullptr, 0, 2 * LATD, 0);
        lstm_gate<<<gdiv(BD * LATD, TPB), blk, 0, stream>>>(Zg, c1,
                                             L1in + LATD, 2 * LATD,   // h1 (dec input + next step)
                                             nullptr, 0,
                                             hfin, BD * LATD);        // last step survives = h_fin[-1]

        // ---- decoder + residual ----
        gemm_bf16<<<gHID, blk, 0, stream>>>(L1in + LATD, 2 * LATD, decW0T, LATD, db0,
                                            nullptr, 0, nullptr, 0, E0, HIDD, LATD, 1);
        gemm_bf16<<<gHID, blk, 0, stream>>>(E0, HIDD, decW1T, HIDD, db1,
                                            nullptr, 0, nullptr, 0, E1, HIDD, HIDD, 1);
        gemm_bf16<<<gOBS, blk, 0, stream>>>(E1, HIDD, decW2T, HIDD, db2,
                                            xt, OBSD,                               // residual x_t
                                            outF + (size_t)t * BD * OBSD, OBSD,     // pred_xs[t]
                                            nullptr, 0, HIDD, 0);
    }
}
